// VisualContrastiveLoss_54125177864431
// MI455X (gfx1250) — matrix operand order repaired and checked by reading, not compiled, not run on hardware
//
#include <hip/hip_runtime.h>
#include <hip/hip_bf16.h>

// ---------------------------------------------------------------------------
// VisualContrastiveLoss on MI455X (gfx1250, wave32, WMMA)
//   loss = mean_i ( logsumexp_j(f_i.f_j/T) - (f_i.f_{label_i})/T ),  label in {0,1}
// Final design: M=16 rows/wave with A (16x512 bf16) resident in 128 VGPRs,
// dual j-tiles (N=32) per iteration sharing A, ~170 VGPRs total so 2-3 waves
// co-reside per SIMD and hide each other's B-fragment load waits (rounds 3-5
// showed the 1-wave/SIMD extended-VGPR variants end in serialization, load
// rematerialization, or scratch spills).
// ---------------------------------------------------------------------------

#define B_DIM 8192
#define D_DIM 512
#define TEMP_INV 14.2857142857142857f           /* 1/0.07 */
#define LOG2E 1.4426950408889634f

typedef __bf16 bf16_t;
typedef __attribute__((ext_vector_type(16))) __bf16 v16bf;
typedef __attribute__((ext_vector_type(8)))  __bf16 v8bf;
typedef __attribute__((ext_vector_type(4)))  __bf16 v4bf;
typedef __attribute__((ext_vector_type(8)))  float  v8f;

// ---------------- Kernel 1: L2-normalize rows (f32) -> bf16 panel ----------
__global__ void vcl_normalize(const float* __restrict__ x,
                              bf16_t* __restrict__ fb) {
    const int row = blockIdx.x;                  // one block (128 thr) per row
    const int t   = threadIdx.x;                 // 128 * 4 = 512 elements
    const float4 v = ((const float4*)(x + (size_t)row * D_DIM))[t];
    float ss = v.x * v.x + v.y * v.y + v.z * v.z + v.w * v.w;
    #pragma unroll
    for (int off = 16; off >= 1; off >>= 1)
        ss += __shfl_xor(ss, off, 32);
    __shared__ float wsum[4];
    if ((t & 31) == 0) wsum[t >> 5] = ss;
    __syncthreads();
    const float tot   = wsum[0] + wsum[1] + wsum[2] + wsum[3];
    const float scale = 1.0f / fmaxf(sqrtf(tot), 1e-12f);
    v4bf o;
    o[0] = (bf16_t)(v.x * scale);
    o[1] = (bf16_t)(v.y * scale);
    o[2] = (bf16_t)(v.z * scale);
    o[3] = (bf16_t)(v.w * scale);
    *(v4bf*)(fb + (size_t)row * D_DIM + t * 4) = o;
}

// ---------------- Kernel 2: faithful label construction --------------------
__global__ void vcl_labels(const int* __restrict__ ids,
                           const int* __restrict__ anchor_p,
                           int* __restrict__ labels, int B) {
    const int i = blockIdx.x * blockDim.x + threadIdx.x;
    if (i >= B) return;
    const int aid      = ids[anchor_p[0]];
    const int sameLast = (ids[B - 1] == aid);
    const int samePrev = (ids[B - 2] == aid);
    const int samei    = (ids[i] == aid);
    int lab = (samei && sameLast) ? 1 : 0;
    if (i == B - 1) lab = (sameLast && samePrev) ? 1 : 0;
    labels[i] = lab;
}

// B fragment (matches A: lane half h holds K = k0 + h*8+{0..7} and
// k0 + 16+h*8+{0..7}) -> two global_load_b128 per fragment.
__device__ __forceinline__ v16bf load_bfrag(const bf16_t* __restrict__ brow,
                                            int k0, int h) {
    const v8bf lo = *(const v8bf*)(brow + k0 + h * 8);
    const v8bf hi = *(const v8bf*)(brow + k0 + 16 + h * 8);
    v16bf b;
    #pragma unroll
    for (int e = 0; e < 8; ++e) { b[e] = lo[e]; b[8 + e] = hi[e]; }
    return b;
}

// ---------------- Kernel 3: fused WMMA GEMM + streaming LSE ----------------
// One wave per 16-row block; A (16x512 bf16) lives in 128 VGPRs for the whole
// 8192-column sweep. Two 16-col tiles per iteration share A; co-resident
// waves on each SIMD hide the B-fragment load latency.
__global__ void __launch_bounds__(128)
vcl_lse_gemm(const bf16_t* __restrict__ fb,
             const int* __restrict__ labels,
             float* __restrict__ rowvals) {
    const int lane = threadIdx.x & 31;
    const int wave = threadIdx.x >> 5;
    const int i0   = (blockIdx.x * 4 + wave) * 16;   // this wave's row block
    const int m    = lane & 15;                      // row / col within tile
    const int h    = lane >> 4;                      // lane half

    // ---- Load A fragments (ISA 16-bit A 16x32 layout):
    //      lane half h holds K = h*8+{0..7} and 16+h*8+{0..7}
    const bf16_t* arow = fb + (size_t)(i0 + m) * D_DIM;
    v16bf a[16];
    #pragma unroll
    for (int kk = 0; kk < 16; ++kk) {
        const int k0 = kk * 32;
        const v8bf lo = *(const v8bf*)(arow + k0 + h * 8);
        const v8bf hi = *(const v8bf*)(arow + k0 + 16 + h * 8);
        #pragma unroll
        for (int e = 0; e < 8; ++e) { a[kk][e] = lo[e]; a[kk][8 + e] = hi[e]; }
    }

    // dot <= 1 (normalized rows)  =>  fixed-max LSE: sum exp((dot-1)/T).
    // exp2-domain: t = dot*K1 - K1, K1 = log2e/T.  Diagonal term = 1 => sm>=1.
    const float K1 = TEMP_INV * LOG2E;
    float sm[8];
    float s0v[8], s1v[8];
    #pragma unroll
    for (int v = 0; v < 8; ++v) sm[v] = 0.0f;

    for (int j0 = 0; j0 < B_DIM; j0 += 32) {
        // Prefetch the two tiles of the next iteration (global_prefetch_b8)
        if (j0 + 32 < B_DIM) {
            __builtin_prefetch(fb + (size_t)(j0 + 32 + m) * D_DIM, 0, 3);
            __builtin_prefetch(fb + (size_t)(j0 + 48 + m) * D_DIM, 0, 3);
        }

        const bf16_t* brow0 = fb + (size_t)(j0 + m) * D_DIM;
        const bf16_t* brow1 = fb + (size_t)(j0 + 16 + m) * D_DIM;

        v8f c0 = {}, c1 = {};
        v16bf b0 = load_bfrag(brow0, 0, h);
        v16bf b1 = load_bfrag(brow1, 0, h);
        #pragma unroll
        for (int kk = 0; kk < 15; ++kk) {
            v16bf nb0 = load_bfrag(brow0, (kk + 1) * 32, h);
            v16bf nb1 = load_bfrag(brow1, (kk + 1) * 32, h);
            c0 = __builtin_amdgcn_wmma_f32_16x16x32_bf16(
                     false, a[kk], false, b0, (short)0, c0, false, false);
            c1 = __builtin_amdgcn_wmma_f32_16x16x32_bf16(
                     false, a[kk], false, b1, (short)0, c1, false, false);
            b0 = nb0; b1 = nb1;
        }
        c0 = __builtin_amdgcn_wmma_f32_16x16x32_bf16(
                 false, a[15], false, b0, (short)0, c0, false, false);
        c1 = __builtin_amdgcn_wmma_f32_16x16x32_bf16(
                 false, a[15], false, b1, (short)0, c1, false, false);

        // C layout: lane holds column n=m, rows v + 8h across 8 regs
        #pragma unroll
        for (int v = 0; v < 8; ++v)
            sm[v] += exp2f(c0[v] * K1 - K1) + exp2f(c1[v] * K1 - K1);

        if (j0 == 0) {  // capture sim[i,0], sim[i,1] (scaled) via bpermute
            #pragma unroll
            for (int v = 0; v < 8; ++v) {
                const float cs = c0[v] * TEMP_INV;
                s0v[v] = __shfl(cs, h * 16 + 0, 32);
                s1v[v] = __shfl(cs, h * 16 + 1, 32);
            }
        }
    }

    // Reduce column-subsets across the 16-lane half (disjoint column cover)
    #pragma unroll
    for (int v = 0; v < 8; ++v) {
        #pragma unroll
        for (int off = 8; off >= 1; off >>= 1)
            sm[v] += __shfl_xor(sm[v], off, 32);
    }

    if (m == 0) {  // lanes 0 (rows 0..7) and 16 (rows 8..15)
        #pragma unroll
        for (int v = 0; v < 8; ++v) {
            const int   r      = i0 + v + 8 * h;
            const float lse    = TEMP_INV + logf(sm[v]);   // = 1/T + ln(sm)
            const float picked = labels[r] ? s1v[v] : s0v[v];
            rowvals[r] = lse - picked;
        }
    }
}

// ---------------- Kernel 4: deterministic mean reduction -------------------
__global__ void vcl_reduce(const float* __restrict__ rowvals,
                           float* __restrict__ out, int B) {
    __shared__ float sh[256];
    float s = 0.0f;
    for (int i = threadIdx.x; i < B; i += 256) s += rowvals[i];
    sh[threadIdx.x] = s;
    __syncthreads();
    #pragma unroll
    for (int off = 128; off >= 1; off >>= 1) {
        if (threadIdx.x < off) sh[threadIdx.x] += sh[threadIdx.x + off];
        __syncthreads();
    }
    if (threadIdx.x == 0) out[0] = sh[0] / (float)B;
}

// ---------------------------------------------------------------------------
extern "C" void kernel_launch(void* const* d_in, const int* in_sizes, int n_in,
                              void* d_out, int out_size, void* d_ws, size_t ws_size,
                              hipStream_t stream) {
    const float* feat   = (const float*)d_in[0];   // [8192,512] f32
    const int*   ids    = (const int*)d_in[1];     // [8192] i32
    const int*   anchor = (const int*)d_in[2];     // [1] i32
    float*       out    = (float*)d_out;           // scalar f32
    const int B = in_sizes[1];                     // 8192

    char*   ws      = (char*)d_ws;
    bf16_t* fb      = (bf16_t*)ws;                                   // 8 MB
    int*    labels  = (int*)(ws + (size_t)B * D_DIM * sizeof(bf16_t));
    float*  rowvals = (float*)(ws + (size_t)B * D_DIM * sizeof(bf16_t)
                                  + (size_t)B * sizeof(int));

    vcl_normalize<<<B, 128, 0, stream>>>(feat, fb);
    vcl_labels<<<(B + 255) / 256, 256, 0, stream>>>(ids, anchor, labels, B);
    vcl_lse_gemm<<<B / 64, 128, 0, stream>>>(fb, labels, rowvals);
    vcl_reduce<<<1, 256, 0, stream>>>(rowvals, out, B);
}
